// KalmanBlock_18356690223141
// MI455X (gfx1250) — matrix operand order repaired and checked by reading, not compiled, not run on hardware
//
#include <hip/hip_runtime.h>
#include <hip/hip_bf16.h>
#include <hip/hip_fp16.h>

typedef __attribute__((ext_vector_type(16))) _Float16 v16h;
typedef __attribute__((ext_vector_type(8)))  _Float16 v8h;
typedef __attribute__((ext_vector_type(8)))  float    v8f;

#define T_SEQ 2048
#define BATCH 8
#define EMB   1024
#define DIN   256
#define SDIM  128
#define MROWS (BATCH * T_SEQ)   // 16384

__device__ __forceinline__ float gelu_exact(float x) {
    return 0.5f * x * (1.0f + erff(x * 0.70710678118654752440f));
}
__device__ __forceinline__ float softplus_f(float x) {
    return (x > 20.0f) ? x : log1pf(expf(x));
}
__device__ __forceinline__ v8h cvt8_f32_f16(const float* __restrict__ p) {
    v8h r;
#pragma unroll
    for (int e = 0; e < 8; ++e) r[e] = (_Float16)p[e];
    return r;
}
__device__ __forceinline__ v16h cat16(v8h lo, v8h hi) {
    return __builtin_shufflevector(lo, hi, 0,1,2,3,4,5,6,7,8,9,10,11,12,13,14,15);
}

// ---------------------------------------------------------------------------
// Pack Wi^T and Wo^T into f16 (B-operand layout: [N][K], contiguous K)
// ---------------------------------------------------------------------------
__global__ void pack_weights(const float* __restrict__ Wi, const float* __restrict__ Wo,
                             _Float16* __restrict__ WiT, _Float16* __restrict__ WoT) {
    int idx = blockIdx.x * blockDim.x + threadIdx.x;
    if (idx < DIN * EMB) {                       // 262144: WiT[n][k] = Wi[k][n]
        int n = idx / EMB, k = idx % EMB;
        WiT[idx] = (_Float16)Wi[k * DIN + n];
    } else {
        int j = idx - DIN * EMB;                 // 131072: WoT[n][k] = Wo[k][n]
        if (j < EMB * SDIM) {
            int n = j / SDIM, k = j % SDIM;
            WoT[j] = (_Float16)Wo[k * EMB + n];
        }
    }
}

// ---------------------------------------------------------------------------
// GEMM1: z = gelu(x @ Wi + bi)   [16384,1024] x [1024,256] -> [16384,256]
// 8 waves per block, each wave computes a 16x64 strip via v_wmma_f32_16x16x32_f16
// ---------------------------------------------------------------------------
__global__ void __launch_bounds__(256)
gemm_in_gelu(const float* __restrict__ x, const _Float16* __restrict__ WiT,
             const float* __restrict__ bi, float* __restrict__ z) {
    const int lane  = threadIdx.x & 31;
    const int wv    = threadIdx.x >> 5;
    const int m0    = blockIdx.x * 128 + wv * 16;
    const int n0    = blockIdx.y * 64;
    const int row   = m0 + (lane & 15);
    const int grp   = lane >> 4;          // 0 or 1
    const int khalf = grp * 16;           // A-layout K offset
    const int kb16  = grp * 16;           // B-layout K offset

    v8f acc0 = {}, acc1 = {}, acc2 = {}, acc3 = {};
    const float* xr = x + (size_t)row * EMB;

    for (int kk = 0; kk < EMB; kk += 32) {
        // A fragment (contiguous K, matches B): lo = K[khalf..+7], hi = K[khalf+8..+7]
        v16h a = cat16(cvt8_f32_f16(xr + kk + khalf),
                       cvt8_f32_f16(xr + kk + 8 + khalf));
#pragma unroll
        for (int nt = 0; nt < 4; ++nt) {
            const _Float16* br = WiT + (size_t)(n0 + nt * 16 + (lane & 15)) * EMB + kk + kb16;
            v16h bf = *(const v16h*)br;   // 16 contiguous K values per lane
            v8f* ap = (nt == 0) ? &acc0 : (nt == 1) ? &acc1 : (nt == 2) ? &acc2 : &acc3;
            *ap = __builtin_amdgcn_wmma_f32_16x16x32_f16(false, a, false, bf,
                                                         (short)0, *ap, false, false);
        }
    }
    const int rbase = m0 + 8 * grp;       // C layout: vgpr v -> row m0 + v + 8*(lane/16)
#pragma unroll
    for (int nt = 0; nt < 4; ++nt) {
        int nc = n0 + nt * 16 + (lane & 15);
        v8f acc = (nt == 0) ? acc0 : (nt == 1) ? acc1 : (nt == 2) ? acc2 : acc3;
        float bb = bi[nc];
#pragma unroll
        for (int v = 0; v < 8; ++v)
            z[(size_t)(rbase + v) * DIN + nc] = gelu_exact(acc[v] + bb);
    }
}

// ---------------------------------------------------------------------------
// GEMM2: out = states @ Wo + bo   [16384,128] x [128,1024] -> [16384,1024]
// ---------------------------------------------------------------------------
__global__ void __launch_bounds__(256)
gemm_out(const float* __restrict__ states, const _Float16* __restrict__ WoT,
         const float* __restrict__ bo, float* __restrict__ out) {
    const int lane  = threadIdx.x & 31;
    const int wv    = threadIdx.x >> 5;
    const int m0    = blockIdx.x * 128 + wv * 16;
    const int n0    = blockIdx.y * 64;
    const int row   = m0 + (lane & 15);
    const int grp   = lane >> 4;
    const int khalf = grp * 16;
    const int kb16  = grp * 16;

    v8f acc0 = {}, acc1 = {}, acc2 = {}, acc3 = {};
    const float* sr = states + (size_t)row * SDIM;

    for (int kk = 0; kk < SDIM; kk += 32) {
        v16h a = cat16(cvt8_f32_f16(sr + kk + khalf),
                       cvt8_f32_f16(sr + kk + 8 + khalf));
#pragma unroll
        for (int nt = 0; nt < 4; ++nt) {
            const _Float16* br = WoT + (size_t)(n0 + nt * 16 + (lane & 15)) * SDIM + kk + kb16;
            v16h bf = *(const v16h*)br;
            v8f* ap = (nt == 0) ? &acc0 : (nt == 1) ? &acc1 : (nt == 2) ? &acc2 : &acc3;
            *ap = __builtin_amdgcn_wmma_f32_16x16x32_f16(false, a, false, bf,
                                                         (short)0, *ap, false, false);
        }
    }
    const int rbase = m0 + 8 * grp;
#pragma unroll
    for (int nt = 0; nt < 4; ++nt) {
        int nc = n0 + nt * 16 + (lane & 15);
        v8f acc = (nt == 0) ? acc0 : (nt == 1) ? acc1 : (nt == 2) ? acc2 : acc3;
        float bb = bo[nc];
#pragma unroll
        for (int v = 0; v < 8; ++v)
            out[(size_t)(rbase + v) * EMB + nc] = acc[v] + bb;
    }
}

// ---------------------------------------------------------------------------
// Sequential Kalman scan. One workgroup per batch item; all weights resident
// in LDS (~265 KB, needs CDNA5's 320 KB WGP LDS). fp32 recurrence.
// ---------------------------------------------------------------------------
__global__ void __launch_bounds__(256)
kalman_scan(const float* __restrict__ z, const float* __restrict__ A,
            const float* __restrict__ H, const float* __restrict__ Ws,
            const float* __restrict__ bs, const float* __restrict__ Q,
            const float* __restrict__ R, float* __restrict__ states,
            float* __restrict__ final_state, float* __restrict__ P_full) {
    extern __shared__ char smem[];
    float* A_s     = (float*)smem;              // 128 x 129 (padded, conflict-free)
    float* Ws_s    = A_s + 128 * 129;           // 256 x 129
    float* state_s = Ws_s + 256 * 129;          // 128
    float* sp_s    = state_s + 128;             // 128
    float* P_s     = sp_s + 128;                // 128
    float* Pp_s    = P_s + 128;                 // 128
    float* sq_s    = Pp_s + 128;                // 128
    float* bs_s    = sq_s + 128;                // 128
    float* inn_s   = bs_s + 128;                // 256
    float* part_s  = inn_s + 256;               // 256
    _Float16* H_s  = (_Float16*)(part_s + 256); // 256 x 130 (halfs)

    const int tid = threadIdx.x;
    const int b   = blockIdx.x;

    for (int idx = tid; idx < SDIM * SDIM; idx += 256)
        A_s[(idx >> 7) * 129 + (idx & 127)] = A[idx];
    for (int idx = tid; idx < DIN * SDIM; idx += 256) {
        int j = idx >> 7, k = idx & 127;
        Ws_s[j * 129 + k] = Ws[idx];
        H_s[j * 130 + k]  = (_Float16)H[idx];
    }
    if (tid < SDIM) {
        sq_s[tid]    = softplus_f(Q[tid]);
        bs_s[tid]    = bs[tid];
        state_s[tid] = 0.0f;
        P_s[tid]     = 1.0f;
    }
    inn_s[tid] = softplus_f(R[tid]);            // stage softplus(R)
    __syncthreads();
    float avgR = 0.0f;
    for (int j = 0; j < DIN; ++j) avgR += inn_s[j];
    avgR *= (1.0f / (float)DIN);
    __syncthreads();

    const int i_lo = tid & 127;                 // output index for split dots
    const int half = tid >> 7;                  // 0/1

    for (int t = 0; t < T_SEQ; ++t) {
        const float* zr = z + ((size_t)b * T_SEQ + t) * DIN;
        float zt = zr[tid];                     // issued early, latency overlapped
        if (t + 1 < T_SEQ)
            __builtin_prefetch(zr + DIN + tid, 0, 1);   // global_prefetch next row

        // state_pred = state @ A^T  (split 128-dot over 2 threads)
        {
            float s = 0.0f;
            const float* ar = A_s + i_lo * 129 + half * 64;
            const float* st = state_s + half * 64;
#pragma unroll 8
            for (int j = 0; j < 64; ++j) s = fmaf(ar[j], st[j], s);
            part_s[tid] = s;
        }
        __syncthreads();
        if (tid < SDIM) {
            sp_s[tid] = part_s[tid] + part_s[tid + 128];
            Pp_s[tid] = P_s[tid] + sq_s[tid];
        }
        __syncthreads();

        // innovation = zt - state_pred @ H^T  (one 128-dot per thread, f16 weights)
        {
            float a = zt;
            const _Float16* hr = H_s + tid * 130;
#pragma unroll 8
            for (int k = 0; k < SDIM; ++k) a = fmaf(-(float)hr[k], sp_s[k], a);
            inn_s[tid] = a;
        }
        __syncthreads();

        // state_update = innovation @ Ws + bs  (split 256-dot over 2 threads)
        {
            float s = 0.0f;
            const int j0 = half * 128;
#pragma unroll 8
            for (int j = 0; j < 128; ++j) s = fmaf(inn_s[j0 + j], Ws_s[(j0 + j) * 129 + i_lo], s);
            part_s[tid] = s;
        }
        __syncthreads();

        if (tid < SDIM) {
            float su = part_s[tid] + part_s[tid + 128] + bs_s[tid];
            float ap = 0.0f;
            for (int j = 0; j < SDIM; ++j) ap += Pp_s[j];   // broadcast reads
            ap *= (1.0f / (float)SDIM);
            float Sinv = 1.0f / (ap + avgR + 1e-6f);
            float Kg   = Pp_s[tid] * Sinv;
            float ns   = sp_s[tid] + Kg * su;
            state_s[tid] = ns;
            states[((size_t)b * T_SEQ + t) * SDIM + tid] = ns;
            P_s[tid] = fmaxf(Pp_s[tid] * (1.0f - Kg), 1e-6f);
        }
        __syncthreads();
    }

    if (tid < SDIM) final_state[b * SDIM + tid] = state_s[tid];
    for (int idx = tid; idx < SDIM * SDIM; idx += 256) {
        int r_ = idx >> 7, c_ = idx & 127;
        P_full[(size_t)b * SDIM * SDIM + idx] = (r_ == c_) ? P_s[r_] : 0.0f;
    }
}

// ---------------------------------------------------------------------------
extern "C" void kernel_launch(void* const* d_in, const int* in_sizes, int n_in,
                              void* d_out, int out_size, void* d_ws, size_t ws_size,
                              hipStream_t stream) {
    const float* x  = (const float*)d_in[0];
    const float* Wi = (const float*)d_in[1];
    const float* bi = (const float*)d_in[2];
    const float* Ws = (const float*)d_in[3];
    const float* bs = (const float*)d_in[4];
    const float* Wo = (const float*)d_in[5];
    const float* bo = (const float*)d_in[6];
    const float* A  = (const float*)d_in[7];
    const float* H  = (const float*)d_in[8];
    const float* Q  = (const float*)d_in[9];
    const float* R  = (const float*)d_in[10];

    float* out    = (float*)d_out;                      // [8,2048,1024]
    float* out_fs = out + (size_t)MROWS * EMB;          // [8,128]
    float* out_P  = out_fs + BATCH * SDIM;              // [8,128,128]

    float*    z_ws      = (float*)d_ws;                             // 16 MB
    float*    states_ws = z_ws + (size_t)MROWS * DIN;               // 8 MB
    _Float16* WiT       = (_Float16*)(states_ws + (size_t)MROWS * SDIM);
    _Float16* WoT       = WiT + (size_t)DIN * EMB;

    pack_weights<<<1536, 256, 0, stream>>>(Wi, Wo, WiT, WoT);
    gemm_in_gelu<<<dim3(128, 4), 256, 0, stream>>>(x, WiT, bi, z_ws);

    size_t smem = (size_t)(128 * 129 + 256 * 129 + 6 * 128 + 2 * 256) * 4
                + (size_t)256 * 130 * 2;                // 269,824 B (< 320 KB WGP LDS)
    kalman_scan<<<BATCH, 256, smem, stream>>>(z_ws, A, H, Ws, bs, Q, R,
                                              states_ws, out_fs, out_P);

    gemm_out<<<dim3(128, 16), 256, 0, stream>>>(states_ws, WoT, bo, out);
}
